// Attention_48687749267849
// MI455X (gfx1250) — matrix operand order repaired and checked by reading, compiled, not run on hardware
//
#include <hip/hip_runtime.h>
#include <hip/hip_bf16.h>
#include <math.h>

#define DIM   384
#define HEADS 12
#define HD    32          // head dim
#define NPOS  784         // 28*28
#define NB    16          // batch
#define MP    800         // m padded to multiple of 32
#define NT16  49          // 784/16

typedef __bf16 bf16_t;
typedef __attribute__((ext_vector_type(16))) __bf16 v16bf;
typedef __attribute__((ext_vector_type(8)))  __bf16 v8bf;
typedef __attribute__((ext_vector_type(8)))  float  v8f;

// ---------------------------------------------------------------- WMMA core
__device__ inline v8f wmma_bf16(v16bf a, v16bf b, v8f c) {
  // (neg_a, A, neg_b, B, c_mod, C, reuse_a, reuse_b)
  return __builtin_amdgcn_wmma_f32_16x16x32_bf16(false, a, false, b, (short)0, c,
                                                 false, false);
}

// A fragment (16x32, M x K), source row-major [M][K], lda in elements.
// lane L: row = L%16, K = (L/16)*8 .. +7  and  (L/16)*8+16 .. +23
__device__ inline v16bf load_a16x32(const bf16_t* base, int lda, int lane) {
  const int row = lane & 15;
  const int kb  = (lane >> 4) << 3;
  const bf16_t* p = base + (size_t)row * lda + kb;
  v16bf a;
  ((v8bf*)&a)[0] = *(const v8bf*)(p);
  ((v8bf*)&a)[1] = *(const v8bf*)(p + 16);
  return a;
}

// B fragment (32x16, K x N), source stored transposed row-major [N][K], ldb elems.
// lane L: col = L%16, K = (L/16)*8 .. +7  and  (L/16)*8+16 .. +23
__device__ inline v16bf load_b32x16(const bf16_t* base, int ldb, int lane) {
  const int col = lane & 15;
  const int kb  = (lane >> 4) << 3;
  const bf16_t* p = base + (size_t)col * ldb + kb;
  v16bf b;
  ((v8bf*)&b)[0] = *(const v8bf*)(p);
  ((v8bf*)&b)[1] = *(const v8bf*)(p + 16);
  return b;
}

// ---------------------------------------------------------------- utilities
__global__ void k_f32_to_bf16(const float* __restrict__ src, bf16_t* __restrict__ dst, int n) {
  int i = blockIdx.x * 256 + threadIdx.x;
  if (i < n) dst[i] = (bf16_t)src[i];
}

// x (B, C=384, N=784) fp32 -> xT (B, N, C) bf16   (B operand for QKV GEMM)
__global__ void k_xT(const float* __restrict__ x, bf16_t* __restrict__ xT) {
  int i = blockIdx.x * 256 + threadIdx.x;
  if (i >= NB * NPOS * DIM) return;
  int c = i % DIM; int t = i / DIM; int n = t % NPOS; int b = t / NPOS;
  xT[i] = (bf16_t)x[((size_t)b * DIM + c) * NPOS + n];
}

// zero the m-padding of V
__global__ void k_vpad(bf16_t* __restrict__ V) {
  int i = blockIdx.x * 256 + threadIdx.x;
  if (i >= NB * HEADS * HD * (MP - NPOS)) return;
  int p = i % (MP - NPOS); int t = i / (MP - NPOS);
  int dv = t % HD; t /= HD; int h = t % HEADS; int b = t / HEADS;
  V[(((size_t)(b * HEADS + h) * HD) + dv) * MP + NPOS + p] = (bf16_t)0.0f;
}

// ---------------------------------------------------------------- QKV GEMM
// y[o][n] = sum_c W[o][c] * x[c][n]  ; A=W row-major, B=xT[n][c]
// kvmode==0: dstQK = Qt[b][h][n][d]       (o = h*32+d)
// kvmode==1: o<384 -> Kc[b][h][m][d] ; o>=384 -> V[b][h][dv][m]
__global__ void k_qkv(const bf16_t* __restrict__ W, const bf16_t* __restrict__ xT,
                      const float* __restrict__ bias,
                      bf16_t* __restrict__ dstQK, bf16_t* __restrict__ dstV,
                      int kvmode, int total) {
  int tile = blockIdx.x * (blockDim.x >> 5) + (threadIdx.x >> 5);
  if (tile >= total) return;
  int lane = threadIdx.x & 31;
  const int OT = kvmode ? (2 * DIM / 16) : (DIM / 16);
  int nt = tile % NT16; int t = tile / NT16; int ot = t % OT; int b = t / OT;

  const bf16_t* wbase = W + (size_t)ot * 16 * DIM;
  const bf16_t* xbase = xT + ((size_t)b * NPOS + nt * 16) * DIM;
  v8f c = {};
  for (int kb = 0; kb < DIM; kb += 32) {
    v16bf a  = load_a16x32(wbase + kb, DIM, lane);
    v16bf bm = load_b32x16(xbase + kb, DIM, lane);
    c = wmma_bf16(a, bm, c);
  }
  int col = lane & 15, r0 = (lane >> 4) << 3;
  int n = nt * 16 + col;
  for (int j = 0; j < 8; ++j) {
    int o = ot * 16 + r0 + j;
    float v = c[j] + bias[o];
    if (o < DIM) {                        // Q (mode 0) or K (mode 1)
      int h = o >> 5, d = o & 31;
      dstQK[(((size_t)(b * HEADS + h) * NPOS) + n) * HD + d] = (bf16_t)v;
    } else {                              // V, stored [dv][m]
      int o2 = o - DIM; int h = o2 >> 5, dv = o2 & 31;
      dstV[(((size_t)(b * HEADS + h) * HD) + dv) * MP + n] = (bf16_t)v;
    }
  }
}

// ---------------------------------------------------------------- scores
// S[h][n][m] = scale * sum_d Qt[n][d]*Kc[m][d]   (K=32 -> one WMMA per tile)
// S stored bf16 (logits are O(1); conv re-accumulates in fp32)
__global__ void k_score(const bf16_t* __restrict__ Qt, const bf16_t* __restrict__ Kc,
                        bf16_t* __restrict__ S, float scale, int b, int total) {
  int tile = blockIdx.x * (blockDim.x >> 5) + (threadIdx.x >> 5);
  if (tile >= total) return;
  int lane = threadIdx.x & 31;
  int mt = tile % NT16; int t = tile / NT16; int nt = t % NT16; int h = t / NT16;
  const bf16_t* qb = Qt + (((size_t)(b * HEADS + h) * NPOS) + nt * 16) * HD;
  const bf16_t* kb = Kc + (((size_t)(b * HEADS + h) * NPOS) + mt * 16) * HD;
  v16bf a  = load_a16x32(qb, HD, lane);
  v16bf bm = load_b32x16(kb, HD, lane);
  v8f c = {};
  c = wmma_bf16(a, bm, c);
  int col = lane & 15, r0 = (lane >> 4) << 3;
  bf16_t* sp = S + ((size_t)h * NPOS + nt * 16) * NPOS + mt * 16 + col;
  for (int j = 0; j < 8; ++j)
    sp[(size_t)(r0 + j) * NPOS] = (bf16_t)(c[j] * scale);
}

// ---------------------------------------------------------------- fused refine:
// 3x3 talking-heads conv + shortcut + bre + rel-pos bias -> softmax -> 12x12
// head mixer -> P (bf16, m padded). One block per query row n.
#define CH  196    // m-chunk (784/4)
#define CHP 198    // + halo
__global__ __launch_bounds__(256)
void k_refine(const bf16_t* __restrict__ S, const float* __restrict__ Wre,
              const float* __restrict__ bre, const float* __restrict__ rpb,
              const int* __restrict__ rel, const float* __restrict__ Wrp,
              const float* __restrict__ brp, bf16_t* __restrict__ P) {
  __shared__ float A[HEADS * NPOS];            // 37.6 KB refined logits / probs
  __shared__ float sh[3 * HEADS * CHP];        // 28.5 KB conv halo stage
  __shared__ float wre_s[HEADS * HEADS * 9];   // 5.2 KB
  __shared__ float wrp_s[HEADS * HEADS];
  __shared__ float red[256];
  int n = blockIdx.x, tid = threadIdx.x;
  for (int i = tid; i < HEADS * HEADS * 9; i += 256) wre_s[i] = Wre[i];
  for (int i = tid; i < HEADS * HEADS; i += 256) wrp_s[i] = Wrp[i];

  // ---- conv + bias, chunked over m, accumulating into A
  for (int cch = 0; cch < 4; ++cch) {
    int m0 = cch * CH;
    for (int i = tid; i < 3 * HEADS * CHP; i += 256) {
      int mm = i % CHP; int t = i / CHP; int hh = t % HEADS; int r = t / HEADS;
      int gn = n + r - 1, gm = m0 + mm - 1;
      float v = 0.0f;
      if (gn >= 0 && gn < NPOS && gm >= 0 && gm < NPOS)
        v = (float)S[((size_t)hh * NPOS + gn) * NPOS + gm];
      sh[i] = v;
    }
    __syncthreads();
    for (int i = tid; i < HEADS * CH; i += 256) {
      int ml = i % CH; int ho = i / CH; int m = m0 + ml;
      float acc = sh[(1 * HEADS + ho) * CHP + ml + 1]                  // shortcut
                + bre[ho]
                + rpb[(size_t)rel[(size_t)n * NPOS + m] * HEADS + ho]; // rel-pos bias
      for (int hi = 0; hi < HEADS; ++hi) {
        const float* w = &wre_s[(ho * HEADS + hi) * 9];
        for (int r = 0; r < 3; ++r) {
          const float* srow = &sh[(r * HEADS + hi) * CHP + ml];
          acc += w[r * 3 + 0] * srow[0] + w[r * 3 + 1] * srow[1] + w[r * 3 + 2] * srow[2];
        }
      }
      A[ho * NPOS + m] = acc;
    }
    __syncthreads();   // protect sh before next chunk overwrites it
  }

  // ---- softmax per head over m
  for (int h = 0; h < HEADS; ++h) {
    float mx = -1e30f;
    for (int m = tid; m < NPOS; m += 256) mx = fmaxf(mx, A[h * NPOS + m]);
    red[tid] = mx; __syncthreads();
    for (int s = 128; s > 0; s >>= 1) {
      if (tid < s) red[tid] = fmaxf(red[tid], red[tid + s]);
      __syncthreads();
    }
    mx = red[0]; __syncthreads();
    float sm = 0.0f;
    for (int m = tid; m < NPOS; m += 256) {
      float e = __expf(A[h * NPOS + m] - mx);
      A[h * NPOS + m] = e; sm += e;
    }
    red[tid] = sm; __syncthreads();
    for (int s = 128; s > 0; s >>= 1) {
      if (tid < s) red[tid] += red[tid + s];
      __syncthreads();
    }
    float inv = 1.0f / red[0]; __syncthreads();
    for (int m = tid; m < NPOS; m += 256) A[h * NPOS + m] *= inv;
    __syncthreads();
  }

  // ---- P = A + (Wrp @ A + brp), bf16, m padded to MP
  for (int i = tid; i < HEADS * NPOS; i += 256) {
    int m = i % NPOS; int h = i / NPOS;
    float p = brp[h];
    for (int h2 = 0; h2 < HEADS; ++h2) p += wrp_s[h * HEADS + h2] * A[h2 * NPOS + m];
    P[((size_t)h * NPOS + n) * MP + m] = (bf16_t)(A[i] + p);
  }
  for (int i = tid; i < HEADS * (MP - NPOS); i += 256) {
    int h = i / (MP - NPOS); int mp = NPOS + i % (MP - NPOS);
    P[((size_t)h * NPOS + n) * MP + mp] = (bf16_t)0.0f;
  }
}

// ---------------------------------------------------------------- attn @ V
// O1[n][h*32+dv] = sum_m P[h][n][m] * V[h][dv][m]; both dv halves share A.
__global__ void k_av(const bf16_t* __restrict__ P, const bf16_t* __restrict__ V,
                     bf16_t* __restrict__ O1, int b, int total) {
  int tile = blockIdx.x * (blockDim.x >> 5) + (threadIdx.x >> 5);
  if (tile >= total) return;           // total = HEADS * NT16
  int lane = threadIdx.x & 31;
  int nt = tile % NT16; int h = tile / NT16;
  const bf16_t* pb = P + ((size_t)h * NPOS + nt * 16) * MP;
  const bf16_t* v0 = V + (((size_t)(b * HEADS + h) * HD) + 0) * MP;
  const bf16_t* v1 = v0 + (size_t)16 * MP;
  v8f c0 = {}, c1 = {};
  for (int kb = 0; kb < MP; kb += 32) {
    v16bf a  = load_a16x32(pb + kb, MP, lane);
    v16bf b0 = load_b32x16(v0 + kb, MP, lane);
    v16bf b1 = load_b32x16(v1 + kb, MP, lane);
    c0 = wmma_bf16(a, b0, c0);
    c1 = wmma_bf16(a, b1, c1);
  }
  int col = lane & 15, r0 = (lane >> 4) << 3;
  for (int j = 0; j < 8; ++j) {
    int nrow = nt * 16 + r0 + j;
    O1[(size_t)nrow * DIM + h * HD + col]      = (bf16_t)c0[j];
    O1[(size_t)nrow * DIM + h * HD + 16 + col] = (bf16_t)c1[j];
  }
}

// ---------------------------------------------------------------- output projection
__global__ void k_out(const bf16_t* __restrict__ WoB, const bf16_t* __restrict__ O1,
                      const float* __restrict__ bo, float* __restrict__ out,
                      int b, int total) {
  int tile = blockIdx.x * (blockDim.x >> 5) + (threadIdx.x >> 5);
  if (tile >= total) return;           // total = (DIM/16) * NT16
  int lane = threadIdx.x & 31;
  int nt = tile % NT16; int ot = tile / NT16;
  const bf16_t* wb = WoB + (size_t)ot * 16 * DIM;
  const bf16_t* xb = O1 + (size_t)nt * 16 * DIM;
  v8f c = {};
  for (int kb = 0; kb < DIM; kb += 32) {
    v16bf a  = load_a16x32(wb + kb, DIM, lane);
    v16bf bm = load_b32x16(xb + kb, DIM, lane);
    c = wmma_bf16(a, bm, c);
  }
  int col = lane & 15, r0 = (lane >> 4) << 3;
  int n = nt * 16 + col;
  for (int j = 0; j < 8; ++j) {
    int o = ot * 16 + r0 + j;
    out[((size_t)(b * DIM + o)) * NPOS + n] = c[j] + bo[o];
  }
}

// ---------------------------------------------------------------- host
extern "C" void kernel_launch(void* const* d_in, const int* in_sizes, int n_in,
                              void* d_out, int out_size, void* d_ws, size_t ws_size,
                              hipStream_t stream) {
  (void)in_sizes; (void)n_in; (void)out_size; (void)ws_size;
  const float* x   = (const float*)d_in[0];
  const float* Wq  = (const float*)d_in[1];
  const float* bq  = (const float*)d_in[2];
  const float* Wkv = (const float*)d_in[3];
  const float* bkv = (const float*)d_in[4];
  const float* Wre = (const float*)d_in[5];
  const float* bre = (const float*)d_in[6];
  const float* Wrp = (const float*)d_in[7];
  const float* brp = (const float*)d_in[8];
  const float* rpb = (const float*)d_in[9];
  const float* Wo  = (const float*)d_in[10];
  const float* bo  = (const float*)d_in[11];
  const int*   rel = (const int*)d_in[12];
  float* out = (float*)d_out;

  char* base = (char*)d_ws;
  size_t off = 0;
  auto alloc = [&](size_t bytes) -> void* {
    void* p = base + off;
    off = (off + bytes + 255) & ~(size_t)255;
    return p;
  };
  bf16_t* WqB  = (bf16_t*)alloc((size_t)DIM * DIM * 2);
  bf16_t* WkvB = (bf16_t*)alloc((size_t)2 * DIM * DIM * 2);
  bf16_t* WoB  = (bf16_t*)alloc((size_t)DIM * DIM * 2);
  bf16_t* xT   = (bf16_t*)alloc((size_t)NB * NPOS * DIM * 2);
  bf16_t* Qt   = (bf16_t*)alloc((size_t)NB * HEADS * NPOS * HD * 2);
  bf16_t* Kc   = (bf16_t*)alloc((size_t)NB * HEADS * NPOS * HD * 2);
  bf16_t* Vb   = (bf16_t*)alloc((size_t)NB * HEADS * HD * MP * 2);
  bf16_t* S    = (bf16_t*)alloc((size_t)HEADS * NPOS * NPOS * 2);  // per-batch reuse
  bf16_t* P    = (bf16_t*)alloc((size_t)HEADS * NPOS * MP * 2);    // per-batch reuse
  bf16_t* O1   = (bf16_t*)alloc((size_t)NPOS * DIM * 2);           // per-batch reuse

  // weight conversion + input transpose + V pad zero
  k_f32_to_bf16<<<(DIM * DIM + 255) / 256, 256, 0, stream>>>(Wq, WqB, DIM * DIM);
  k_f32_to_bf16<<<(2 * DIM * DIM + 255) / 256, 256, 0, stream>>>(Wkv, WkvB, 2 * DIM * DIM);
  k_f32_to_bf16<<<(DIM * DIM + 255) / 256, 256, 0, stream>>>(Wo, WoB, DIM * DIM);
  k_xT<<<(NB * NPOS * DIM + 255) / 256, 256, 0, stream>>>(x, xT);
  k_vpad<<<(NB * HEADS * HD * (MP - NPOS) + 255) / 256, 256, 0, stream>>>(Vb);

  // QKV projections (all batches)
  {
    int tilesQ = NB * (DIM / 16) * NT16;
    k_qkv<<<(tilesQ + 3) / 4, 128, 0, stream>>>(WqB, xT, bq, Qt, Vb, 0, tilesQ);
    int tilesKV = NB * (2 * DIM / 16) * NT16;
    k_qkv<<<(tilesKV + 3) / 4, 128, 0, stream>>>(WkvB, xT, bkv, Kc, Vb, 1, tilesKV);
  }

  const float scale = 1.0f / sqrtf((float)HD);

  // attention pipeline, per batch (stream-ordered; S/P/O1 reused)
  for (int b = 0; b < NB; ++b) {
    int tilesS = HEADS * NT16 * NT16;
    k_score<<<(tilesS + 3) / 4, 128, 0, stream>>>(Qt, Kc, S, scale, b, tilesS);
    k_refine<<<NPOS, 256, 0, stream>>>(S, Wre, bre, rpb, rel, Wrp, brp, P);
    int tilesAV = HEADS * NT16;
    k_av<<<(tilesAV + 3) / 4, 128, 0, stream>>>(P, Vb, O1, b, tilesAV);
    int tilesO = (DIM / 16) * NT16;
    k_out<<<(tilesO + 3) / 4, 128, 0, stream>>>(WoB, O1, bo, out, b, tilesO);
  }
}
